// IndRNN_14697378087441
// MI455X (gfx1250) — matrix operand order repaired and checked by reading, not compiled, not run on hardware
//
#include <hip/hip_runtime.h>

// ---------------------------------------------------------------------------
// IndRNN forward for MI455X (gfx1250, wave32, WMMA).
//   Phase 1: Wx = X @ W + bias  via bf16x3-split V_WMMA_F32_16X16X32_BF16
//            (fp32-accurate), double-buffered LDS pipeline, written directly
//            into d_out[0 : T*B*H].
//   Phase 2: in-place diagonal scan h_t = tanh(wx_t + h_{t-1}*u) with a
//            hardware-transcendental tanh; final state appended at the end.
// ---------------------------------------------------------------------------

typedef __attribute__((ext_vector_type(16))) __bf16 v16bf;
typedef __attribute__((ext_vector_type(8)))  __bf16 v8bf;
typedef __attribute__((ext_vector_type(8)))  float  v8f;
typedef unsigned short u16;

#define T_DIM 2048
#define B_DIM 32
#define I_DIM 512
#define H_DIM 1024
#define M_DIM (T_DIM * B_DIM)   // 65536 rows of the input-projection GEMM

#define BM 128                  // block tile M
#define BN 128                  // block tile N
#define BK 32                   // K step == WMMA K for bf16
#define KITERS (I_DIM / BK)     // 16

__device__ __forceinline__ u16 f32_bf16_rne(float x) {
    unsigned u = __float_as_uint(x);
    u += 0x7FFFu + ((u >> 16) & 1u);      // round-to-nearest-even
    return (u16)(u >> 16);
}
__device__ __forceinline__ float bf16_f32(u16 b) {
    return __uint_as_float(((unsigned)b) << 16);
}
__device__ __forceinline__ v16bf cat8(v8bf lo, v8bf hi) {
    v16bf r;
#pragma unroll
    for (int i = 0; i < 8; ++i) { r[i] = lo[i]; r[i + 8] = hi[i]; }
    return r;
}

// ---------------------------------------------------------------------------
// Phase 1: GEMM.  grid = (M/BM)*(N/BN) = 512*8 = 4096 blocks, 256 thr (8 waves)
// ---------------------------------------------------------------------------
__global__ __launch_bounds__(256) void indrnn_gemm_wmma(
    const float* __restrict__ X,     // [M_DIM, I_DIM]
    const float* __restrict__ W,     // [I_DIM, H_DIM]
    const float* __restrict__ bias,  // [H_DIM]
    float* __restrict__ out)         // [M_DIM, H_DIM]  (== d_out Wx region)
{
    // Ping-pong bf16 hi/lo tiles.  A: [m][k] row-major, B transposed [n][k].
    __shared__ u16 sAh[2][BM * BK];
    __shared__ u16 sAl[2][BM * BK];
    __shared__ u16 sBh[2][BN * BK];
    __shared__ u16 sBl[2][BN * BK];

    const int tid  = threadIdx.x;
    const int lane = tid & 31;
    const int wave = tid >> 5;
    const int half = lane >> 4;   // 0: lanes 0-15, 1: lanes 16-31
    const int lm   = lane & 15;

    const int nBlk = H_DIM / BN;                      // 8
    const int m0   = (blockIdx.x / nBlk) * BM;
    const int n0   = (blockIdx.x % nBlk) * BN;

    const int waveM = wave >> 2;  // 0..1 -> 64-row strip
    const int waveN = wave & 3;   // 0..3 -> 32-col strip

    const v8f vzero = {0.f, 0.f, 0.f, 0.f, 0.f, 0.f, 0.f, 0.f};
    v8f acc[4][2];
#pragma unroll
    for (int tm = 0; tm < 4; ++tm)
#pragma unroll
        for (int tn = 0; tn < 2; ++tn) acc[tm][tn] = vzero;

    // Register staging for the software pipeline (16 A + 16 B floats/thread).
    float ra[16], rb[16];

    auto load_global = [&](int k0) {
#pragma unroll
        for (int j = 0; j < 16; ++j) {
            int idx = tid + j * 256;
            int r = idx >> 5, c = idx & 31;
            ra[j] = X[(size_t)(m0 + r) * I_DIM + (k0 + c)];
        }
#pragma unroll
        for (int j = 0; j < 16; ++j) {
            int idx = tid + j * 256;
            int kr = idx >> 7, nc = idx & 127;
            rb[j] = W[(size_t)(k0 + kr) * H_DIM + (n0 + nc)];
        }
    };
    auto store_tiles = [&](int buf) {
#pragma unroll
        for (int j = 0; j < 16; ++j) {
            int idx = tid + j * 256;
            int r = idx >> 5, c = idx & 31;
            u16 h = f32_bf16_rne(ra[j]);
            u16 l = f32_bf16_rne(ra[j] - bf16_f32(h));
            sAh[buf][r * BK + c] = h;
            sAl[buf][r * BK + c] = l;
        }
#pragma unroll
        for (int j = 0; j < 16; ++j) {
            int idx = tid + j * 256;
            int kr = idx >> 7, nc = idx & 127;
            u16 h = f32_bf16_rne(rb[j]);
            u16 l = f32_bf16_rne(rb[j] - bf16_f32(h));
            sBh[buf][nc * BK + kr] = h;
            sBl[buf][nc * BK + kr] = l;
        }
    };

    load_global(0);
    int buf = 0;

    for (int it = 0; it < KITERS; ++it) {
        // Convert + commit the staged tile (waits on its global loads here).
        store_tiles(buf);
        __syncthreads();

        // Kick off next tile's global loads; they fly during the WMMA phase.
        if (it + 1 < KITERS) load_global((it + 1) * BK);

        // ---- assemble fragments from LDS (gfx1250 wave32 layouts) ----
        // A (16-bit 16x32): half 0 -> K [0..7],[16..23]; half 1 -> [8..15],[24..31]
        v16bf ah[4], al[4];
#pragma unroll
        for (int tm = 0; tm < 4; ++tm) {
            int m = waveM * 64 + tm * 16 + lm;
            const u16* bh_ = &sAh[buf][m * BK + half * 8];
            ah[tm] = cat8(*(const v8bf*)(bh_), *(const v8bf*)(bh_ + 16));
            const u16* bl_ = &sAl[buf][m * BK + half * 8];
            al[tm] = cat8(*(const v8bf*)(bl_), *(const v8bf*)(bl_ + 16));
        }
        // B (16-bit 32x16): lane = col, half 0 -> K [0..7],[16..23]; half 1 -> [8..15],[24..31];
        // B stored transposed [n][k] so the K values are contiguous.
        v16bf bhf[2], blf[2];
#pragma unroll
        for (int tn = 0; tn < 2; ++tn) {
            int n = waveN * 32 + tn * 16 + lm;
            const u16* bh_ = &sBh[buf][n * BK + half * 8];
            bhf[tn] = cat8(*(const v8bf*)(bh_), *(const v8bf*)(bh_ + 16));
            const u16* bl_ = &sBl[buf][n * BK + half * 8];
            blf[tn] = cat8(*(const v8bf*)(bl_), *(const v8bf*)(bl_ + 16));
        }

        // ---- bf16x3 WMMA: lo*hi + hi*lo + hi*hi  (~fp32 accuracy) ----
#pragma unroll
        for (int tm = 0; tm < 4; ++tm)
#pragma unroll
            for (int tn = 0; tn < 2; ++tn) {
                acc[tm][tn] = __builtin_amdgcn_wmma_f32_16x16x32_bf16(
                    false, al[tm], false, bhf[tn], (short)0, acc[tm][tn], false, false);
                acc[tm][tn] = __builtin_amdgcn_wmma_f32_16x16x32_bf16(
                    false, ah[tm], false, blf[tn], (short)0, acc[tm][tn], false, false);
                acc[tm][tn] = __builtin_amdgcn_wmma_f32_16x16x32_bf16(
                    false, ah[tm], false, bhf[tn], (short)0, acc[tm][tn], false, false);
            }
        buf ^= 1;
    }

    // ---- store C + bias.  C layout: lane = col (lm), half picks M+8, vgpr r = row
#pragma unroll
    for (int tn = 0; tn < 2; ++tn) {
        int col = n0 + waveN * 32 + tn * 16 + lm;
        float bv = bias[col];
#pragma unroll
        for (int tm = 0; tm < 4; ++tm) {
            int row0 = m0 + waveM * 64 + tm * 16 + half * 8;
            v8f c = acc[tm][tn];
#pragma unroll
            for (int r = 0; r < 8; ++r)
                out[(size_t)(row0 + r) * H_DIM + col] = c[r] + bv;
        }
    }
}

// ---------------------------------------------------------------------------
// Phase 2: diagonal recurrence, in place over d_out.
// Latency-bound (only B*H = 32768 threads), so keep the dependent chain to
// one v_exp_f32 + one v_rcp_f32 + a few VALU per timestep.
// ---------------------------------------------------------------------------
__device__ __forceinline__ float fast_tanh(float x) {
    // tanh(x) = sign(x) * (1 - 2 / (exp2(2*log2(e)*|x|) + 1))
    float a = __builtin_fabsf(x);
    float e = __builtin_amdgcn_exp2f(a * 2.8853900817779268f);
    float r = 1.0f - 2.0f * __builtin_amdgcn_rcpf(e + 1.0f);
    return __builtin_copysignf(r, x);
}

__global__ __launch_bounds__(256) void indrnn_scan(
    float* __restrict__ out, const float* __restrict__ u)
{
    const int idx = blockIdx.x * blockDim.x + threadIdx.x;   // 0 .. B*H-1
    const float uu = u[idx & (H_DIM - 1)];
    const size_t stride = (size_t)B_DIM * H_DIM;             // 32768
    float* p = out + idx;

    float h  = 0.f;
    float wx = p[0];
#pragma unroll 4
    for (int t = 0; t < T_DIM; ++t) {
        float nxt = (t + 1 < T_DIM) ? p[(size_t)(t + 1) * stride] : 0.f;
        h = fast_tanh(fmaf(h, uu, wx));
        p[(size_t)t * stride] = h;
        wx = nxt;
    }
    out[(size_t)T_DIM * stride + idx] = h;                   // final state
}

// ---------------------------------------------------------------------------
extern "C" void kernel_launch(void* const* d_in, const int* in_sizes, int n_in,
                              void* d_out, int out_size, void* d_ws, size_t ws_size,
                              hipStream_t stream) {
    (void)in_sizes; (void)n_in; (void)d_ws; (void)ws_size; (void)out_size;
    const float* X    = (const float*)d_in[0];   // [T,B,I]
    const float* W    = (const float*)d_in[1];   // [I,H]
    const float* u    = (const float*)d_in[2];   // [H]
    const float* bias = (const float*)d_in[3];   // [H]
    float* out = (float*)d_out;                  // [T*B*H] output + [B*H] state

    indrnn_gemm_wmma<<<dim3((M_DIM / BM) * (H_DIM / BN)), dim3(256), 0, stream>>>(
        X, W, bias, out);
    indrnn_scan<<<dim3((B_DIM * H_DIM) / 256), dim3(256), 0, stream>>>(out, u);
}
